// SelfAttentionWide_27539330302315
// MI455X (gfx1250) — matrix operand order repaired and checked by reading, not compiled, not run on hardware
//
#include <hip/hip_runtime.h>
#include <hip/hip_bf16.h>
#include <cmath>

typedef _Float16 f16;
typedef __attribute__((ext_vector_type(16))) _Float16 v16h;
typedef __attribute__((ext_vector_type(8)))  _Float16 v8h;
typedef __attribute__((ext_vector_type(4)))  _Float16 v4h;
typedef __attribute__((ext_vector_type(8)))  float    v8f;

// ---------------------------------------------------------------------------
// gfx1250 async copy: 16B global -> LDS, tracked by ASYNCcnt (ISA 15.18.3 / ch.8)
// ---------------------------------------------------------------------------
__device__ __forceinline__ void async_ld16(unsigned lds_off, const void* g)
{
  asm volatile("global_load_async_to_lds_b128 %0, %1, off"
               :: "v"(lds_off), "v"(g) : "memory");
}
__device__ __forceinline__ void async_wait0()
{
  asm volatile("s_wait_asynccnt 0x0" ::: "memory");
}

// ---------------------------------------------------------------------------
// Batched tiled WMMA GEMM for gfx1250 (wave32).
//   C[z][M,N] = alpha * A[z][M,K] * B[z]  (+ bias[n])
//   z = blockIdx.z, per-operand offset = (z/H)*s?b + (z%H)*s?h  (elements)
//   A: row-major (lda). B: TRANS_B ? [N,K] row-major : [K,N] row-major.
//   Assumes M%128==0, N%128==0, K%64==0 (true for all GEMMs here).
// Block = 256 threads = 8 waves; tile 128x128x64; wave owns 32x64
// (2x4 of 16x16 -> 8 v8f accumulators, 16 WMMAs per K-block per wave).
// f16 operands are staged with global_load_async_to_lds_b128 (no VGPR bounce);
// f32 operands are converted through VGPRs (v_cvt_pk_f16_f32 + ds_store_b128).
// ---------------------------------------------------------------------------
#define BM 128
#define BN 128
#define BKK 64
#define LDSS 80   // 64 + 16 halfs pad; row = 160B (16B aligned, 2-way conflict max)

template <typename TA, typename TB, typename TC, bool TRANS_B, bool BIAS>
__global__ __launch_bounds__(256)
void wmma_gemm_kernel(const TA* __restrict__ A, int lda,
                      const TB* __restrict__ B, int ldb,
                      TC* __restrict__ C, int ldc,
                      int M, int N, int K, float alpha,
                      const float* __restrict__ bias,
                      int H, size_t sAb, size_t sAh,
                      size_t sBb, size_t sBh, size_t sCb, size_t sCh)
{
  __shared__ __align__(16) f16 As[BM * LDSS];
  __shared__ __align__(16) f16 Bs[BN * LDSS];

  constexpr bool ASYNC_A = (sizeof(TA) == 2);
  constexpr bool ASYNC_B = TRANS_B && (sizeof(TB) == 2);

  const int z  = blockIdx.z;
  const int zb = z / H, zh = z % H;
  const TA* Ab = A + (size_t)zb * sAb + (size_t)zh * sAh;
  const TB* Bb = B + (size_t)zb * sBb + (size_t)zh * sBh;
  TC*       Cb = C + (size_t)zb * sCb + (size_t)zh * sCh;

  const int tid  = threadIdx.x;
  const int lane = tid & 31;
  const int wave = tid >> 5;
  const int row0 = (wave >> 1) * 32;
  const int col0 = (wave & 1) * 64;

  const int m0 = blockIdx.y * BM;
  const int n0 = blockIdx.x * BN;

  // A staging: thread -> row = tid>>1, 32 contiguous halfs at k = (tid&1)*32
  const int aRow = tid >> 1;
  const int aK   = (tid & 1) * 32;
  // NN-B staging: 4x4 micro-tiles (coalesced reads, packed v4h LDS stores)
  const int tk = (tid >> 4) * 4;   // k group 0..60
  const int tn = (tid & 15) * 4;   // n group 0..60 (+ nn in {0,64})

  v8f acc[2][4];
#pragma unroll
  for (int i = 0; i < 2; ++i)
#pragma unroll
    for (int j = 0; j < 4; ++j) {
      v8f zz = {};
      acc[i][j] = zz;
    }

  const int lh     = lane >> 4;  // lane half
  const int kbaseA = lh * 16;    // A frag K packing: {0..15}/{16..31}
  const int kbaseB = lh * 16;    // B frag K packing: {0..15}/{16..31}

  for (int kk = 0; kk < K; kk += BKK) {
    // ---- stage A tile: As[m][k]
    if constexpr (ASYNC_A) {
      const char* src = (const char*)(Ab + (size_t)(m0 + aRow) * lda + (kk + aK));
      const unsigned dst = (unsigned)(size_t)&As[aRow * LDSS + aK];
#pragma unroll
      for (int i = 0; i < 4; ++i) async_ld16(dst + 16 * i, src + 16 * i);
    } else {
      const TA* src = Ab + (size_t)(m0 + aRow) * lda + (kk + aK);
      f16* dst = &As[aRow * LDSS + aK];
#pragma unroll
      for (int i = 0; i < 32; ++i) dst[i] = (f16)src[i];
      if (kk + BKK < K)
        __builtin_prefetch(src + BKK, 0, 3);
    }
    // ---- stage B tile into Bs[n][k] (K-major per column)
    if constexpr (TRANS_B) {
      const int bN = tid >> 1;
      const int bK = (tid & 1) * 32;
      if constexpr (ASYNC_B) {
        const char* src = (const char*)(Bb + (size_t)(n0 + bN) * ldb + (kk + bK));
        const unsigned dst = (unsigned)(size_t)&Bs[bN * LDSS + bK];
#pragma unroll
        for (int i = 0; i < 4; ++i) async_ld16(dst + 16 * i, src + 16 * i);
      } else {
        const TB* src = Bb + (size_t)(n0 + bN) * ldb + (kk + bK);
        f16* dst = &Bs[bN * LDSS + bK];
#pragma unroll
        for (int i = 0; i < 32; ++i) dst[i] = (f16)src[i];
      }
    } else {
#pragma unroll
      for (int nn = 0; nn < BN; nn += 64) {
        f16 tmp[4][4];
#pragma unroll
        for (int kq = 0; kq < 4; ++kq) {
          const TB* s = Bb + (size_t)(kk + tk + kq) * ldb + (n0 + tn + nn);
#pragma unroll
          for (int nq = 0; nq < 4; ++nq) tmp[kq][nq] = (f16)s[nq];
        }
#pragma unroll
        for (int nq = 0; nq < 4; ++nq) {
          v4h p = {tmp[0][nq], tmp[1][nq], tmp[2][nq], tmp[3][nq]};
          *reinterpret_cast<v4h*>(&Bs[(tn + nn + nq) * LDSS + tk]) = p;
        }
      }
      if (kk + BKK < K)
        __builtin_prefetch(Bb + (size_t)(kk + BKK + tk) * ldb + (n0 + tn), 0, 3);
    }

    if constexpr (ASYNC_A || ASYNC_B) async_wait0();
    __syncthreads();

    // ---- 16 WMMAs per K-block (two 32-deep sub-steps)
#pragma unroll
    for (int ks = 0; ks < BKK; ks += 32) {
      const int kA = ks + kbaseA;
      const int kB = ks + kbaseB;
      v16h afrag[2];
#pragma unroll
      for (int i = 0; i < 2; ++i) {
        const int r = row0 + i * 16 + (lane & 15);
        v8h lo = *reinterpret_cast<const v8h*>(&As[r * LDSS + kA]);
        v8h hi = *reinterpret_cast<const v8h*>(&As[r * LDSS + kA + 8]);
        afrag[i] = __builtin_shufflevector(lo, hi, 0,1,2,3,4,5,6,7,8,9,10,11,12,13,14,15);
      }
      v16h bfrag[4];
#pragma unroll
      for (int j = 0; j < 4; ++j) {
        const int c = col0 + j * 16 + (lane & 15);
        v8h lo = *reinterpret_cast<const v8h*>(&Bs[c * LDSS + kB]);
        v8h hi = *reinterpret_cast<const v8h*>(&Bs[c * LDSS + kB + 8]);
        bfrag[j] = __builtin_shufflevector(lo, hi, 0,1,2,3,4,5,6,7,8,9,10,11,12,13,14,15);
      }
#pragma unroll
      for (int i = 0; i < 2; ++i)
#pragma unroll
        for (int j = 0; j < 4; ++j)
          acc[i][j] = __builtin_amdgcn_wmma_f32_16x16x32_f16(
              false, afrag[i], false, bfrag[j], (short)0, acc[i][j], false, false);
    }
    __syncthreads();
  }

  // ---- epilogue: D layout: VGPR r -> M = r + 8*(lane>>4); lane&15 -> N
#pragma unroll
  for (int i = 0; i < 2; ++i)
#pragma unroll
    for (int j = 0; j < 4; ++j) {
      const int n     = n0 + col0 + j * 16 + (lane & 15);
      const int mbase = m0 + row0 + i * 16 + 8 * lh;
      float bb = 0.0f;
      if constexpr (BIAS) bb = bias[n];
      v8f a = acc[i][j];
#pragma unroll
      for (int r = 0; r < 8; ++r) {
        float v = a[r] * alpha + bb;
        Cb[(size_t)(mbase + r) * ldc + n] = (TC)v;
      }
    }
}

// ---------------------------------------------------------------------------
// In-place row softmax on f16 scores; grid = (rows, batches), 2048 cols.
// ---------------------------------------------------------------------------
__global__ __launch_bounds__(256)
void softmax_rows_f16_kernel(f16* __restrict__ S, int ncols)
{
  __shared__ float red[256];
  const int tid = threadIdx.x;
  f16* row = S + ((size_t)blockIdx.y * gridDim.x + blockIdx.x) * ncols;

  float x[8];
  float m = -1e30f;
#pragma unroll
  for (int j = 0; j < 8; ++j) {
    x[j] = (float)row[tid + j * 256];
    m = fmaxf(m, x[j]);
  }
  red[tid] = m;
  __syncthreads();
  for (int s = 128; s > 0; s >>= 1) {
    if (tid < s) red[tid] = fmaxf(red[tid], red[tid + s]);
    __syncthreads();
  }
  m = red[0];
  __syncthreads();

  float sum = 0.0f;
#pragma unroll
  for (int j = 0; j < 8; ++j) {
    x[j] = __expf(x[j] - m);
    sum += x[j];
  }
  red[tid] = sum;
  __syncthreads();
  for (int s = 128; s > 0; s >>= 1) {
    if (tid < s) red[tid] += red[tid + s];
    __syncthreads();
  }
  const float inv = 1.0f / red[0];
#pragma unroll
  for (int j = 0; j < 8; ++j) row[tid + j * 256] = (f16)(x[j] * inv);
}

__global__ void cvt_f32_f16_kernel(const float* __restrict__ in, f16* __restrict__ out, int n)
{
  int i = blockIdx.x * 256 + threadIdx.x;
  if (i < n) out[i] = (f16)in[i];
}

// ---------------------------------------------------------------------------
extern "C" void kernel_launch(void* const* d_in, const int* in_sizes, int n_in,
                              void* d_out, int out_size, void* d_ws, size_t ws_size,
                              hipStream_t stream)
{
  const float* x  = (const float*)d_in[0];
  const float* Wq = (const float*)d_in[1];
  const float* Wk = (const float*)d_in[2];
  const float* Wv = (const float*)d_in[3];
  const float* Wu = (const float*)d_in[4];
  const float* bu = (const float*)d_in[5];

  const int b = 4, t = 2048, e = 768, h = 8;
  const int BT = b * t;   // 8192
  const int HE = h * e;   // 6144

  // workspace (f16): q16 | k16 | v16 | Wu16 | Sb[32][2048][2048]
  // ao16 aliases q16 (q is dead after the scores GEMM; stream order serializes).
  const size_t sz_slab = (size_t)BT * HE;
  f16* q16  = (f16*)d_ws;
  f16* k16  = q16 + sz_slab;
  f16* v16  = k16 + sz_slab;
  f16* Wu16 = v16 + sz_slab;
  f16* Sb   = Wu16 + (size_t)HE * e;   // 32 * t * t f16
  f16* ao16 = q16;                     // alias

  const float alpha_qk = 1.0f / powf((float)e, 0.25f);
  dim3 blk(256);

  // --- Q/K/V projections: X[8192x768] * W[768x6144] -> f16 slabs (f32 staging)
  {
    dim3 grid(HE / BN, BT / BM, 1);
    wmma_gemm_kernel<float, float, f16, false, false><<<grid, blk, 0, stream>>>(
        x, e, Wq, HE, q16, HE, BT, HE, e, alpha_qk, nullptr, 1, 0, 0, 0, 0, 0, 0);
    wmma_gemm_kernel<float, float, f16, false, false><<<grid, blk, 0, stream>>>(
        x, e, Wk, HE, k16, HE, BT, HE, e, alpha_qk, nullptr, 1, 0, 0, 0, 0, 0, 0);
    wmma_gemm_kernel<float, float, f16, false, false><<<grid, blk, 0, stream>>>(
        x, e, Wv, HE, v16, HE, BT, HE, e, 1.0f, nullptr, 1, 0, 0, 0, 0, 0, 0);
  }

  // --- convert Wu once
  {
    const int n = HE * e;
    cvt_f32_f16_kernel<<<dim3((n + 255) / 256), blk, 0, stream>>>(Wu, Wu16, n);
  }

  // --- scores for ALL (b,h): S[z][t,s] = sum_e q*k (NT, async f16 staging)
  {
    dim3 grid(t / BN, t / BM, b * h);
    wmma_gemm_kernel<f16, f16, f16, true, false><<<grid, blk, 0, stream>>>(
        q16, HE, k16, HE, Sb, t, t, t, e, 1.0f, nullptr,
        h, (size_t)t * HE, (size_t)e,          // A batch strides
        (size_t)t * HE, (size_t)e,             // B batch strides
        (size_t)h * t * t, (size_t)t * t);     // C batch strides
  }

  // --- softmax in place over all 32*2048 rows
  softmax_rows_f16_kernel<<<dim3(t, b * h), blk, 0, stream>>>(Sb, t);

  // --- O[z][t,e] = P * v (NN; A async f16, B micro-tiled) -> ao16 (aliases q16)
  {
    dim3 grid(e / BN, t / BM, b * h);
    wmma_gemm_kernel<f16, f16, f16, false, false><<<grid, blk, 0, stream>>>(
        Sb, t, v16, HE, ao16, HE, t, e, t, 1.0f, nullptr,
        h, (size_t)h * t * t, (size_t)t * t,
        (size_t)t * HE, (size_t)e,
        (size_t)t * HE, (size_t)e);
  }

  // --- output projection: ao16[8192x6144] * Wu16[6144x768] + bu -> f32 out
  {
    dim3 grid(e / BN, BT / BM, 1);
    wmma_gemm_kernel<f16, f16, float, false, true><<<grid, blk, 0, stream>>>(
        ao16, HE, Wu16, e, (float*)d_out, e, BT, e, HE, 1.0f, bu,
        1, 0, 0, 0, 0, 0, 0);
  }
}
